// MultiHeadAttention2D_18210661335452
// MI455X (gfx1250) — matrix operand order repaired and checked by reading, not compiled, not run on hardware
//
#include <hip/hip_runtime.h>
#include <hip/hip_bf16.h>
#include <math.h>

// ---------------------------------------------------------------------------
// CDNA5 / gfx1250 implementation: all GEMMs through v_wmma_f32_16x16x32_bf16.
// ---------------------------------------------------------------------------

typedef __attribute__((ext_vector_type(16))) __bf16 v16bf;
typedef __attribute__((ext_vector_type(8)))  float  v8f;
typedef __attribute__((ext_vector_type(8)))  __bf16 bh8;

#define LEAKY_ALPHA 0.2f
#define NEG_BIG     (-1.0e9f)
#define SCALE_QK    0.125f            // 1/sqrt(64)

__device__ __forceinline__ bh8 bh8_zero() {
  bh8 z;
#pragma unroll
  for (int j = 0; j < 8; ++j) z[j] = (__bf16)0.0f;
  return z;
}

__device__ __forceinline__ void acc_zero(v8f& a) {
#pragma unroll
  for (int j = 0; j < 8; ++j) a[j] = 0.0f;
}

__device__ __forceinline__ void set8(v16bf& f, int base, bh8 t) {
#pragma unroll
  for (int j = 0; j < 8; ++j) f[base + j] = t[j];
}

// A fragment (16x32 bf16), tile stored row-major in LDS with given stride.
// Lane L holds row r0+(L&15); K halves: kh..kh+7 and kh+16..kh+23, kh=k0+8*(L>>4).
__device__ __forceinline__ v16bf frag_a_lds(const __bf16* p, int stride,
                                            int r0, int k0, int lane) {
  const __bf16* row = p + (r0 + (lane & 15)) * stride;
  const int kh = k0 + ((lane >> 4) << 3);
  v16bf f;
  set8(f, 0, *(const bh8*)(row + kh));
  set8(f, 8, *(const bh8*)(row + kh + 16));
  return f;
}

// B fragment (32x16 bf16) from B^T tile stored [n][k] row-major in LDS.
// Lane L holds col n0+(L&15); K halves: kb..kb+7 and kb+16..kb+23, kb=k0+8*(L>>4).
__device__ __forceinline__ v16bf frag_b_lds(const __bf16* p, int stride,
                                            int n0, int k0, int lane) {
  const __bf16* row = p + (n0 + (lane & 15)) * stride;
  const int kb = k0 + ((lane >> 4) << 3);
  v16bf f;
  set8(f, 0, *(const bh8*)(row + kb));
  set8(f, 8, *(const bh8*)(row + kb + 16));
  return f;
}

// A fragment built from fp32 LDS (score buffer), converted to bf16 on the fly.
__device__ __forceinline__ v16bf frag_a_f32(const float* S, int stride,
                                            int r0, int k0, int lane) {
  const float* row = S + (r0 + (lane & 15)) * stride;
  const int kh = k0 + ((lane >> 4) << 3);
  v16bf f;
#pragma unroll
  for (int j = 0; j < 8; ++j) {
    f[j]     = (__bf16)row[kh + j];
    f[8 + j] = (__bf16)row[kh + 16 + j];
  }
  return f;
}

__device__ __forceinline__ v8f wmma_bf16(v16bf a, v16bf b, v8f c) {
  return __builtin_amdgcn_wmma_f32_16x16x32_bf16(false, a, false, b,
                                                 (short)0, c, false, false);
}

// ---------------------------------------------------------------------------
// Prep kernels: fp32 -> bf16 conversion / concat / weight transpose
// ---------------------------------------------------------------------------

__global__ void k_cvt_bf16(__bf16* __restrict__ dst, const float* __restrict__ src,
                           long n) {
  long i = (long)blockIdx.x * blockDim.x + threadIdx.x;
  if (i < n) dst[i] = (__bf16)src[i];
}

// dst[r*2048 + c] = c < 1024 ? s1[r*1024+c] : s2[r*1024 + c-1024]
__global__ void k_concat2_bf16(__bf16* __restrict__ dst,
                               const float* __restrict__ s1,
                               const float* __restrict__ s2, long rows) {
  long i = (long)blockIdx.x * blockDim.x + threadIdx.x;
  if (i >= rows * 2048) return;
  long r = i >> 11;
  int  c = (int)(i & 2047);
  float v = (c < 1024) ? s1[r * 1024 + c] : s2[r * 1024 + (c - 1024)];
  dst[i] = (__bf16)v;
}

// Wt[n*K + k] = W[k*N + n]  (transpose + convert), N = 1024
__global__ void k_weight_t(__bf16* __restrict__ dst, const float* __restrict__ W,
                           int K) {
  long i = (long)blockIdx.x * blockDim.x + threadIdx.x;
  if (i >= (long)K * 1024) return;
  int n = (int)(i / K);
  int k = (int)(i - (long)n * K);
  dst[i] = (__bf16)W[(long)k * 1024 + n];
}

// Wt[n*2048 + k] = k<1024 ? W1[k*1024+n] : W2[(k-1024)*1024+n]
__global__ void k_weight_t2(__bf16* __restrict__ dst,
                            const float* __restrict__ W1,
                            const float* __restrict__ W2) {
  long i = (long)blockIdx.x * blockDim.x + threadIdx.x;
  if (i >= 1024L * 2048) return;
  int n = (int)(i >> 11);
  int k = (int)(i & 2047);
  float v = (k < 1024) ? W1[(long)k * 1024 + n] : W2[(long)(k - 1024) * 1024 + n];
  dst[i] = (__bf16)v;
}

// ---------------------------------------------------------------------------
// Generic WMMA GEMM:  C[M,N] = act(A[M,K] @ Wt[N,K]^T + bias0 (+bias1))
// Block tile 128x128, BK=32, 256 threads = 8 waves, wave tile 32x64.
// ---------------------------------------------------------------------------
__global__ __launch_bounds__(256)
void gemm_bf16_kernel(const __bf16* __restrict__ A, const __bf16* __restrict__ Wt,
                      const float* __restrict__ bias0, const float* __restrict__ bias1,
                      int M, int N, int K, int act,
                      __bf16* __restrict__ outb, float* __restrict__ outf) {
  __shared__ __bf16 As[128 * 32];
  __shared__ __bf16 Bs[128 * 32];

  const int tid  = threadIdx.x;
  const int lane = tid & 31;
  const int wave = tid >> 5;
  const int m0 = blockIdx.y * 128;
  const int n0 = blockIdx.x * 128;
  const int wm = (wave & 3) * 32;
  const int wn = (wave >> 2) * 64;

  const int srow = tid >> 1;          // 0..127
  const int sseg = (tid & 1) * 16;    // 0 or 16

  bh8 ra0 = bh8_zero(), ra1 = bh8_zero(), rb0 = bh8_zero(), rb1 = bh8_zero();

  auto ldg = [&](int k0) {
    const int gr = m0 + srow;
    if (gr < M) {
      const __bf16* p = A + (size_t)gr * K + k0 + sseg;
      ra0 = *(const bh8*)p;
      ra1 = *(const bh8*)(p + 8);
      __builtin_prefetch(p + 32, 0, 0);     // global_prefetch_b8 next tile
    } else {
      ra0 = bh8_zero(); ra1 = bh8_zero();
    }
    const __bf16* q = Wt + (size_t)(n0 + srow) * K + k0 + sseg;
    rb0 = *(const bh8*)q;
    rb1 = *(const bh8*)(q + 8);
  };

  v8f acc[2][4];
#pragma unroll
  for (int mi = 0; mi < 2; ++mi)
#pragma unroll
    for (int ni = 0; ni < 4; ++ni) acc_zero(acc[mi][ni]);

  const int nk = K >> 5;
  ldg(0);
  for (int s = 0; s < nk; ++s) {
    __syncthreads();
    *(bh8*)&As[srow * 32 + sseg]     = ra0;
    *(bh8*)&As[srow * 32 + sseg + 8] = ra1;
    *(bh8*)&Bs[srow * 32 + sseg]     = rb0;
    *(bh8*)&Bs[srow * 32 + sseg + 8] = rb1;
    __syncthreads();
    if (s + 1 < nk) ldg((s + 1) << 5);      // overlap next global loads

    v16bf af[2], bfm[4];
#pragma unroll
    for (int mi = 0; mi < 2; ++mi) af[mi] = frag_a_lds(As, 32, wm + mi * 16, 0, lane);
#pragma unroll
    for (int ni = 0; ni < 4; ++ni) bfm[ni] = frag_b_lds(Bs, 32, wn + ni * 16, 0, lane);
#pragma unroll
    for (int mi = 0; mi < 2; ++mi)
#pragma unroll
      for (int ni = 0; ni < 4; ++ni)
        acc[mi][ni] = wmma_bf16(af[mi], bfm[ni], acc[mi][ni]);
  }

  // Epilogue: bias + activation, store bf16 and/or fp32.
#pragma unroll
  for (int mi = 0; mi < 2; ++mi) {
#pragma unroll
    for (int ni = 0; ni < 4; ++ni) {
      const int gc = n0 + wn + ni * 16 + (lane & 15);
      float bv = bias0[gc] + (bias1 ? bias1[gc] : 0.0f);
#pragma unroll
      for (int i = 0; i < 8; ++i) {
        const int row = wm + mi * 16 + i + ((lane >> 4) << 3);
        const int gr  = m0 + row;
        if (gr < M) {
          float v = acc[mi][ni][i] + bv;
          if (act) v = (v > 0.0f) ? v : LEAKY_ALPHA * v;
          if (outb) outb[(size_t)gr * N + gc] = (__bf16)v;
          if (outf) outf[(size_t)gr * N + gc] = v;
        }
      }
    }
  }
}

// ---------------------------------------------------------------------------
// Attention kernel: one workgroup per (b, h, 64-query tile).
// LDS: S[64][1024] fp32 scores (256KB, only possible with 320KB CDNA5 LDS),
// Aq[64][64] bf16 query tile, Bk[128][64] bf16 key/value staging.
// ---------------------------------------------------------------------------
__global__ __launch_bounds__(256)
void attn_kernel(const __bf16* __restrict__ Qp, const __bf16* __restrict__ Kp,
                 const __bf16* __restrict__ Vp, const float* __restrict__ mask,
                 float* __restrict__ attw, __bf16* __restrict__ ctx) {
  extern __shared__ char smem[];
  float*  S   = (float*)smem;                                   // 64*1024 f32
  __bf16* Aq  = (__bf16*)(smem + 64 * 1024 * sizeof(float));    // 64*64
  __bf16* Bk  = (__bf16*)((char*)Aq + 64 * 64 * sizeof(__bf16));// 128*64 (reused 64*128)
  float*  red = (float*)((char*)Bk + 128 * 64 * sizeof(__bf16));// 256
  float*  Mr  = red + 256;                                      // 64
  float*  Li  = Mr + 64;                                        // 64

  const int tid  = threadIdx.x;
  const int lane = tid & 31;
  const int wave = tid >> 5;
  const int qt = blockIdx.x, h = blockIdx.y, b = blockIdx.z;
  const int q0 = qt * 64;

  // ---- load 64x64 query tile: rows are q positions q0+r+1 (the [1:,...] slice)
  {
    const int r = tid >> 2, seg = (tid & 3) * 16;
    const __bf16* p = Qp + ((size_t)b * 1025 + (q0 + r + 1)) * 1024 + h * 64 + seg;
    *(bh8*)&Aq[r * 64 + seg]     = *(const bh8*)p;
    *(bh8*)&Aq[r * 64 + seg + 8] = *(const bh8*)(p + 8);
  }

  const int qm = (wave & 3) * 16;       // wave's 16-row query strip
  const int kn = (wave >> 2) * 64;      // wave's 64-col key strip (within 128 chunk)

  // ---- phase 1: scores (relu*scale + mask) into LDS, 8 chunks of 128 keys
  for (int c = 0; c < 8; ++c) {
    __syncthreads();
    {
      const int n = tid >> 1, seg = (tid & 1) * 32;
      const __bf16* p = Kp + ((size_t)b * 1025 + (c * 128 + n)) * 1024 + h * 64 + seg;
      *(bh8*)&Bk[n * 64 + seg]      = *(const bh8*)p;
      *(bh8*)&Bk[n * 64 + seg + 8]  = *(const bh8*)(p + 8);
      *(bh8*)&Bk[n * 64 + seg + 16] = *(const bh8*)(p + 16);
      *(bh8*)&Bk[n * 64 + seg + 24] = *(const bh8*)(p + 24);
    }
    __syncthreads();

    v8f acc[4];
#pragma unroll
    for (int t = 0; t < 4; ++t) acc_zero(acc[t]);
#pragma unroll
    for (int k0 = 0; k0 < 64; k0 += 32) {
      v16bf a = frag_a_lds(Aq, 64, qm, k0, lane);
#pragma unroll
      for (int t = 0; t < 4; ++t) {
        v16bf bb = frag_b_lds(Bk, 64, kn + t * 16, k0, lane);
        acc[t] = wmma_bf16(a, bb, acc[t]);
      }
    }
#pragma unroll
    for (int t = 0; t < 4; ++t) {
      const int colG = c * 128 + kn + t * 16 + (lane & 15);
      const float* mcol = mask + (size_t)b * 1024 * 1024 + colG;
#pragma unroll
      for (int i = 0; i < 8; ++i) {
        const int row = qm + i + ((lane >> 4) << 3);
        const float mv = mcol[(size_t)(q0 + row) * 1024];
        S[row * 1024 + colG] = fmaxf(acc[t][i] * SCALE_QK, 0.0f) + mv * NEG_BIG;
      }
    }
  }
  __syncthreads();

  // ---- phase 2: exact row softmax in LDS (4 threads per row)
  {
    const int r = tid >> 2, p = tid & 3;
    float* Sr = S + r * 1024 + p * 256;
    float m = -3.0e38f;
    for (int j = 0; j < 256; ++j) m = fmaxf(m, Sr[j]);
    red[tid] = m;
    __syncthreads();
    if (p == 0)
      Mr[r] = fmaxf(fmaxf(red[r * 4], red[r * 4 + 1]),
                    fmaxf(red[r * 4 + 2], red[r * 4 + 3]));
    __syncthreads();
    const float mm = Mr[r];
    float l = 0.0f;
    for (int j = 0; j < 256; ++j) {
      float e = __expf(Sr[j] - mm);
      Sr[j] = e;                        // keep unnormalized exp in LDS
      l += e;
    }
    red[tid] = l;
    __syncthreads();
    if (p == 0)
      Li[r] = 1.0f / (red[r * 4] + red[r * 4 + 1] + red[r * 4 + 2] + red[r * 4 + 3]);
    __syncthreads();
  }

  // ---- phase 2c: coalesced streaming write of normalized att_weights (written once)
  {
    const size_t base = (((size_t)b * 16 + h) * 1024 + q0) * 1024;
    for (int rr = 0; rr < 64; ++rr) {
      const float inv = Li[rr];
      const float* Sr = S + rr * 1024 + tid * 4;
      float4 v = make_float4(Sr[0] * inv, Sr[1] * inv, Sr[2] * inv, Sr[3] * inv);
      *(float4*)&attw[base + (size_t)rr * 1024 + tid * 4] = v;
    }
  }

  // ---- phase 3: ctx = (exp_unnorm @ V_h) * (1/l), WMMA reading A from LDS fp32
  const int nt0 = (wave >> 2) * 2;      // wave owns 2 of the 4 col tiles? (8 waves x 2 = 16 tiles)
  v8f oac[2];
  acc_zero(oac[0]);
  acc_zero(oac[1]);
  for (int c = 0; c < 8; ++c) {
    __syncthreads();
    // stage V chunk transposed: BkT[d][kj], coalesced global reads
    for (int idx = tid; idx < 128 * 64; idx += 256) {
      const int kj = idx >> 6, d = idx & 63;
      Bk[d * 128 + kj] = Vp[((size_t)b * 1024 + c * 128 + kj) * 1024 + h * 64 + d];
    }
    __syncthreads();
#pragma unroll
    for (int ks = 0; ks < 4; ++ks) {
      v16bf a = frag_a_f32(S, 1024, qm, c * 128 + ks * 32, lane);
#pragma unroll
      for (int t = 0; t < 2; ++t) {
        v16bf bb = frag_b_lds(Bk, 128, (nt0 + t) * 16, ks * 32, lane);
        oac[t] = wmma_bf16(a, bb, oac[t]);
      }
    }
  }
#pragma unroll
  for (int t = 0; t < 2; ++t) {
    const int col = (nt0 + t) * 16 + (lane & 15);
#pragma unroll
    for (int i = 0; i < 8; ++i) {
      const int row = qm + i + ((lane >> 4) << 3);
      const float v = oac[t][i] * Li[row];
      ctx[((size_t)b * 1024 + q0 + row) * 1024 + h * 64 + col] = (__bf16)v;
    }
  }
}

// ---------------------------------------------------------------------------
// Host launcher
// ---------------------------------------------------------------------------
extern "C" void kernel_launch(void* const* d_in, const int* in_sizes, int n_in,
                              void* d_out, int out_size, void* d_ws, size_t ws_size,
                              hipStream_t stream) {
  (void)in_sizes; (void)n_in; (void)out_size; (void)ws_size;

  const float* q    = (const float*)d_in[0];
  const float* k    = (const float*)d_in[1];
  const float* v    = (const float*)d_in[2];
  const float* e    = (const float*)d_in[3];
  const float* mask = (const float*)d_in[4];
  const float* Wwq  = (const float*)d_in[5];  const float* bwq  = (const float*)d_in[6];
  const float* Wwq2 = (const float*)d_in[7];  const float* bwq2 = (const float*)d_in[8];
  const float* Wwq3 = (const float*)d_in[9];  const float* bwq3 = (const float*)d_in[10];
  const float* Wwk  = (const float*)d_in[11]; const float* bwk  = (const float*)d_in[12];
  const float* Wwk2 = (const float*)d_in[13]; const float* bwk2 = (const float*)d_in[14];
  const float* Wwk3 = (const float*)d_in[15]; const float* bwk3 = (const float*)d_in[16];
  const float* Wwv  = (const float*)d_in[17]; const float* bwv  = (const float*)d_in[18];
  const float* Wwo  = (const float*)d_in[19]; const float* bwo  = (const float*)d_in[20];

  float* out_f = (float*)d_out;                         // (4,1024,1024)
  float* attw  = out_f + 4ull * 1024 * 1024;            // (4,16,1024,1024)

  // workspace layout (bf16 buffers, all offsets 256B aligned)
  char* w = (char*)d_ws;
  __bf16* Aqe   = (__bf16*)w;               w += 4100ull * 2048 * 2;  // [q|e]
  __bf16* Ake   = (__bf16*)w;               w += 4100ull * 2048 * 2;  // [k|e]
  __bf16* Wcq_t = (__bf16*)w;               w += 1024ull * 2048 * 2;
  __bf16* Wck_t = (__bf16*)w;               w += 1024ull * 2048 * 2;
  __bf16* Wq3_t = (__bf16*)w;               w += 1024ull * 1024 * 2;
  __bf16* Wk3_t = (__bf16*)w;               w += 1024ull * 1024 * 2;
  __bf16* Wv_t  = (__bf16*)w;               w += 1024ull * 1024 * 2;
  __bf16* Wo_t  = (__bf16*)w;               w += 1024ull * 1024 * 2;
  __bf16* Hq    = (__bf16*)w;               w += 4100ull * 1024 * 2;
  __bf16* Hk    = (__bf16*)w;               w += 4100ull * 1024 * 2;
  __bf16* Qp    = (__bf16*)w;               w += 4100ull * 1024 * 2;
  __bf16* Kp    = (__bf16*)w;               w += 4100ull * 1024 * 2;
  __bf16* Vbf   = (__bf16*)w;               w += 4096ull * 1024 * 2;
  __bf16* Vp    = (__bf16*)w;               w += 4096ull * 1024 * 2;
  __bf16* Ctx   = (__bf16*)w;               w += 4096ull * 1024 * 2;

  // ---- prep: conversions / concat / weight transposes
  {
    long n = 4100L * 2048;
    k_concat2_bf16<<<(n + 255) / 256, 256, 0, stream>>>(Aqe, q, e, 4100L);
    k_concat2_bf16<<<(n + 255) / 256, 256, 0, stream>>>(Ake, k, e, 4100L);
  }
  {
    long n = 4096L * 1024;
    k_cvt_bf16<<<(n + 255) / 256, 256, 0, stream>>>(Vbf, v, n);
  }
  {
    long n2 = 1024L * 2048;
    k_weight_t2<<<(n2 + 255) / 256, 256, 0, stream>>>(Wcq_t, Wwq, Wwq2);
    k_weight_t2<<<(n2 + 255) / 256, 256, 0, stream>>>(Wck_t, Wwk, Wwk2);
    long n1 = 1024L * 1024;
    k_weight_t<<<(n1 + 255) / 256, 256, 0, stream>>>(Wq3_t, Wwq3, 1024);
    k_weight_t<<<(n1 + 255) / 256, 256, 0, stream>>>(Wk3_t, Wwk3, 1024);
    k_weight_t<<<(n1 + 255) / 256, 256, 0, stream>>>(Wv_t,  Wwv,  1024);
    k_weight_t<<<(n1 + 255) / 256, 256, 0, stream>>>(Wo_t,  Wwo,  1024);
  }

  const dim3 blk(256);
  const dim3 g4100(1024 / 128, (4100 + 127) / 128);   // 8 x 33
  const dim3 g4096(1024 / 128, 4096 / 128);           // 8 x 32

  // ---- fused first-layer GEMMs (K=2048): leaky(q@Wq + e@Wq2 + bq + bq2)
  gemm_bf16_kernel<<<g4100, blk, 0, stream>>>(Aqe, Wcq_t, bwq, bwq2,
                                              4100, 1024, 2048, 1, Hq, nullptr);
  gemm_bf16_kernel<<<g4100, blk, 0, stream>>>(Ake, Wck_t, bwk, bwk2,
                                              4100, 1024, 2048, 1, Hk, nullptr);
  // ---- second-layer projections
  gemm_bf16_kernel<<<g4100, blk, 0, stream>>>(Hq, Wq3_t, bwq3, nullptr,
                                              4100, 1024, 1024, 0, Qp, nullptr);
  gemm_bf16_kernel<<<g4100, blk, 0, stream>>>(Hk, Wk3_t, bwk3, nullptr,
                                              4100, 1024, 1024, 0, Kp, nullptr);
  // ---- value projection
  gemm_bf16_kernel<<<g4096, blk, 0, stream>>>(Vbf, Wv_t, bwv, nullptr,
                                              4096, 1024, 1024, 0, Vp, nullptr);

  // ---- fused attention: scores + softmax + att_weights write + context GEMM
  {
    const dim3 grid(16, 16, 4);                 // (q-tile, head, batch)
    const size_t smem = 64 * 1024 * sizeof(float)           // S
                      + 64 * 64 * sizeof(__bf16)            // Aq
                      + 128 * 64 * sizeof(__bf16)           // Bk / BvT
                      + (256 + 64 + 64) * sizeof(float);    // reductions
    attn_kernel<<<grid, blk, smem, stream>>>(Qp, Kp, Vp, mask, attw, Ctx);
  }

  // ---- output projection -> fp32 d_out
  gemm_bf16_kernel<<<g4096, blk, 0, stream>>>(Ctx, Wo_t, bwo, nullptr,
                                              4096, 1024, 1024, 0, nullptr, out_f);
}
